// GCNN_Prod_Res_3324304687694
// MI455X (gfx1250) — matrix operand order repaired and checked by reading, not compiled, not run on hardware
//
#include <hip/hip_runtime.h>
#include <hip/hip_bf16.h>

typedef __attribute__((ext_vector_type(16))) __bf16 v16bf;
typedef __attribute__((ext_vector_type(8)))  __bf16 v8bf;
typedef __attribute__((ext_vector_type(4)))  __bf16 v4bf;
typedef __attribute__((ext_vector_type(8)))  float  v8f;

#define CDIM 128
#define KDIM 40
#define EPS  1e-5f
#define TPAD (CDIM + 24)   // 152 bf16 = 304 B row stride: 16B-aligned, conflict-free b128 LDS reads

// ---------------- small utility kernels ----------------

__global__ void k_deg_init(float* __restrict__ deg, int n) {
    int i = blockIdx.x * 256 + threadIdx.x;
    if (i < n) deg[i] = 1.0f;               // self-loop contributes 1
}

__global__ void k_deg_acc(const int* __restrict__ dst, float* __restrict__ deg, int e) {
    int i = blockIdx.x * 256 + threadIdx.x;
    if (i < e) atomicAdd(&deg[dst[i]], 1.0f);
}

__global__ void k_rsqrt_inplace(float* __restrict__ d, int n) {
    int i = blockIdx.x * 256 + threadIdx.x;
    if (i < n) d[i] = rsqrtf(d[i]);         // deg >= 1 guaranteed
}

// Pack W_conv [128][128] f32 into per-lane WMMA B-fragment layout:
// frag index = t*4+s (t = 16-col tile 0..7, s = 32-K step 0..3), then [lane][16] contiguous bf16.
// Lane layout (wave32): n = t*16 + (lane&15), K = 32*s + (lane>>4)*8 + (j&7) + (j>>3)*16.
__global__ void k_pack_wconv(const float* __restrict__ W, __bf16* __restrict__ o) {
    int i = blockIdx.x * 256 + threadIdx.x;           // 8*4*32*16 = 16384 total
    if (i >= 8 * 4 * 32 * 16) return;
    int j    = i & 15;
    int lane = (i >> 4) & 31;
    int s    = (i >> 9) & 3;
    int t    = i >> 11;
    int k = 32 * s + (lane >> 4) * 8 + (j & 7) + (j >> 3) * 16;
    int n = t * 16 + (lane & 15);
    o[i] = (__bf16)W[k * CDIM + n];
}

// ---------------- h = x @ W_conv  (WMMA bf16 -> f32) ----------------
// Block: 256 threads = 8 waves; block covers 128 rows, wave w covers rows [16w,16w+16).
// A frags: two contiguous runs of 8 K-values -> 4x float4 loads + cvt.
// B frags: 32-byte contiguous loads, double-buffered across the column-tile loop
// so next tile's loads overlap the current WMMA chain.

__global__ void __launch_bounds__(256)
k_gemm_conv(const float* __restrict__ x, const __bf16* __restrict__ Wbf,
            float* __restrict__ h, int nrows)
{
    const int tid  = threadIdx.x;
    const int wave = tid >> 5;
    const int lane = tid & 31;
    const int m    = lane & 15;
    const int hi   = lane >> 4;

    const int r0   = blockIdx.x * 128 + wave * 16;
    const int arow = r0 + m;
    const bool rowOk    = (arow < nrows);
    const bool fullTile = (r0 + 16 <= nrows);       // wave-uniform
    const float* xr = x + (size_t)arow * CDIM;

    v16bf a[4];
    if (rowOk) {
#pragma unroll
        for (int s = 0; s < 4; ++s) {
            const float4 f0 = *(const float4*)(xr + 32 * s + hi * 8);
            const float4 f1 = *(const float4*)(xr + 32 * s + hi * 8 + 4);
            const float4 f2 = *(const float4*)(xr + 32 * s + 16 + hi * 8);
            const float4 f3 = *(const float4*)(xr + 32 * s + 16 + hi * 8 + 4);
            v16bf av;
            av[0]  = (__bf16)f0.x; av[1]  = (__bf16)f0.y; av[2]  = (__bf16)f0.z; av[3]  = (__bf16)f0.w;
            av[4]  = (__bf16)f1.x; av[5]  = (__bf16)f1.y; av[6]  = (__bf16)f1.z; av[7]  = (__bf16)f1.w;
            av[8]  = (__bf16)f2.x; av[9]  = (__bf16)f2.y; av[10] = (__bf16)f2.z; av[11] = (__bf16)f2.w;
            av[12] = (__bf16)f3.x; av[13] = (__bf16)f3.y; av[14] = (__bf16)f3.z; av[15] = (__bf16)f3.w;
            a[s] = av;
        }
    } else {
        v16bf z = {};
#pragma unroll
        for (int s = 0; s < 4; ++s) a[s] = z;
    }

    const v16bf* Wf = (const v16bf*)Wbf;   // fragment-layout weights

    // double-buffered B fragments
    v16bf bcur[4], bnext[4];
#pragma unroll
    for (int s = 0; s < 4; ++s) bcur[s] = Wf[s * 32 + lane];

#pragma unroll
    for (int t = 0; t < 8; ++t) {
        if (t < 7) {
#pragma unroll
            for (int s = 0; s < 4; ++s) bnext[s] = Wf[((t + 1) * 4 + s) * 32 + lane];
        }
        v8f c = {};
#pragma unroll
        for (int s = 0; s < 4; ++s) {
            c = __builtin_amdgcn_wmma_f32_16x16x32_bf16(
                    false, a[s], false, bcur[s], (short)0, c, false, false);
        }
        // C/D layout: row = r + 8*hi, col = t*16 + m
        float* hp = h + (size_t)(r0 + hi * 8) * CDIM + t * 16 + m;
        if (fullTile) {
#pragma unroll
            for (int r = 0; r < 8; ++r) hp[r * CDIM] = c[r];   // immediate-offset store clause
        } else {
#pragma unroll
            for (int r = 0; r < 8; ++r)
                if (r0 + hi * 8 + r < nrows) hp[r * CDIM] = c[r];
        }
#pragma unroll
        for (int s = 0; s < 4; ++s) bcur[s] = bnext[s];
    }
}

// ---------------- agg = b_conv + h * dinv^2 (self-loop term) ----------------

__global__ void k_agg_init(const float* __restrict__ h, const float* __restrict__ dinv,
                           const float* __restrict__ b_conv, float* __restrict__ agg, int n)
{
    int i = blockIdx.x * 256 + threadIdx.x;       // one float4 per thread
    int total = n * (CDIM / 4);
    if (i >= total) return;
    int row = i >> 5;
    int c4  = (i & 31) * 4;
    float di = dinv[row];
    float w  = di * di;
    const float4 hv = *(const float4*)(h + (size_t)row * CDIM + c4);
    const float4 bv = *(const float4*)(b_conv + c4);
    float4 o;
    o.x = bv.x + hv.x * w; o.y = bv.y + hv.y * w;
    o.z = bv.z + hv.z * w; o.w = bv.w + hv.w * w;
    *(float4*)(agg + (size_t)row * CDIM + c4) = o;
}

// ---------------- edge scatter: agg[dst] += h[src] * dinv[src]*dinv[dst] ----------------
// 1 wave per edge; each lane handles 4 channels. h/agg are L2-resident (51 MB < 192 MB L2).

__global__ void __launch_bounds__(256)
k_scatter(const int* __restrict__ src, const int* __restrict__ dst,
          const float* __restrict__ h, const float* __restrict__ dinv,
          float* __restrict__ agg, int e)
{
    const int tid  = threadIdx.x;
    const int wave = tid >> 5;
    const int lane = tid & 31;
    const int eidx = blockIdx.x * 8 + wave;
    if (eidx >= e) return;

    const int s = src[eidx];
    const int d = dst[eidx];
    const float w = dinv[s] * dinv[d];

    const float4 hv = *(const float4*)(h + (size_t)s * CDIM + lane * 4);
    float* ap = agg + (size_t)d * CDIM + lane * 4;
    atomicAdd(ap + 0, hv.x * w);
    atomicAdd(ap + 1, hv.y * w);
    atomicAdd(ap + 2, hv.z * w);
    atomicAdd(ap + 3, hv.w * w);
}

// ---------------- fused:  a=relu(agg)*x ; LayerNorm ; out=(ln+x)@W_fc+b_fc ----------------
// Block: 256 threads = 8 waves; 128 rows/block. t = ln+x staged in LDS (bf16, b128-friendly);
// W_fc staged in LDS directly in per-lane fragment layout (3 tiles x 4 ksteps, 40->48 pad).

__global__ void __launch_bounds__(256)
k_ln_fc(const float* __restrict__ agg, const float* __restrict__ x,
        const float* __restrict__ gamma, const float* __restrict__ beta,
        const float* __restrict__ Wfc, const float* __restrict__ bfc,
        float* __restrict__ out, int n)
{
    __shared__ __align__(16) __bf16 ldsT[128][TPAD];
    __shared__ __align__(16) __bf16 ldsWf[3 * 4 * 32 * 16];   // fragment layout, 12 KB

    const int tid  = threadIdx.x;
    const int wave = tid >> 5;
    const int lane = tid & 31;
    const int rowBase = blockIdx.x * 128;

    // stage W_fc into fragment layout (zero-pad cols 40..47)
    for (int idx = tid; idx < 3 * 4 * 32 * 16; idx += 256) {
        int j  = idx & 15;
        int l2 = (idx >> 4) & 31;
        int s  = (idx >> 9) & 3;
        int t  = idx >> 11;
        int k  = 32 * s + (l2 >> 4) * 8 + (j & 7) + (j >> 3) * 16;
        int nn = t * 16 + (l2 & 15);
        ldsWf[idx] = (nn < KDIM) ? (__bf16)Wfc[k * KDIM + nn] : (__bf16)0.0f;
    }

    // phase 1: relu*x + LayerNorm + residual, per-row wave reduction (wave32)
    for (int it = 0; it < 16; ++it) {
        const int lrow = wave + it * 8;
        const int row  = rowBase + lrow;
        v4bf tv = {};
        if (row < n) {
            const float4 av = *(const float4*)(agg + (size_t)row * CDIM + lane * 4);
            const float4 xv = *(const float4*)(x   + (size_t)row * CDIM + lane * 4);
            float a0 = fmaxf(av.x, 0.f) * xv.x;
            float a1 = fmaxf(av.y, 0.f) * xv.y;
            float a2 = fmaxf(av.z, 0.f) * xv.z;
            float a3 = fmaxf(av.w, 0.f) * xv.w;
            float s  = a0 + a1 + a2 + a3;
            float ss = a0 * a0 + a1 * a1 + a2 * a2 + a3 * a3;
#pragma unroll
            for (int off = 16; off > 0; off >>= 1) {
                s  += __shfl_xor(s,  off, 32);
                ss += __shfl_xor(ss, off, 32);
            }
            const float mu   = s * (1.0f / CDIM);
            const float var  = ss * (1.0f / CDIM) - mu * mu;
            const float rstd = rsqrtf(var + EPS);
            const float4 gv = *(const float4*)(gamma + lane * 4);
            const float4 bv = *(const float4*)(beta  + lane * 4);
            tv[0] = (__bf16)((a0 - mu) * rstd * gv.x + bv.x + xv.x);
            tv[1] = (__bf16)((a1 - mu) * rstd * gv.y + bv.y + xv.y);
            tv[2] = (__bf16)((a2 - mu) * rstd * gv.z + bv.z + xv.z);
            tv[3] = (__bf16)((a3 - mu) * rstd * gv.w + bv.w + xv.w);
        }
        *(v4bf*)&ldsT[lrow][lane * 4] = tv;   // 8-byte packed LDS store
    }
    __syncthreads();

    // phase 2: WMMA GEMM (ln+x) @ W_fc
    const int m   = lane & 15;
    const int hi  = lane >> 4;
    const int lr0 = wave * 16;

    const __bf16* tr = &ldsT[lr0 + m][0];
    v16bf a[4];
#pragma unroll
    for (int s = 0; s < 4; ++s) {
        const v8bf lo = *(const v8bf*)(tr + 32 * s + hi * 8);        // 16B ds load
        const v8bf hh = *(const v8bf*)(tr + 32 * s + 16 + hi * 8);   // 16B ds load
        v16bf av;
#pragma unroll
        for (int j = 0; j < 8; ++j) { av[j] = lo[j]; av[j + 8] = hh[j]; }
        a[s] = av;
    }

    const v16bf* Wf = (const v16bf*)ldsWf;
    const bool fullTile = (rowBase + lr0 + 16 <= n);   // wave-uniform

    // double-buffered B fragments from LDS
    v16bf bcur[4], bnext[4];
#pragma unroll
    for (int s = 0; s < 4; ++s) bcur[s] = Wf[s * 32 + lane];

#pragma unroll
    for (int t = 0; t < 3; ++t) {
        if (t < 2) {
#pragma unroll
            for (int s = 0; s < 4; ++s) bnext[s] = Wf[((t + 1) * 4 + s) * 32 + lane];
        }
        v8f c = {};
#pragma unroll
        for (int s = 0; s < 4; ++s) {
            c = __builtin_amdgcn_wmma_f32_16x16x32_bf16(
                    false, a[s], false, bcur[s], (short)0, c, false, false);
        }
        const int col = t * 16 + m;
        if (col < KDIM) {
            const float bb = bfc[col];
            float* op = out + (size_t)(rowBase + lr0 + hi * 8) * KDIM + col;
            if (fullTile) {
#pragma unroll
                for (int r = 0; r < 8; ++r) op[r * KDIM] = c[r] + bb;   // immediate offsets
            } else {
#pragma unroll
                for (int r = 0; r < 8; ++r)
                    if (rowBase + lr0 + hi * 8 + r < n) op[r * KDIM] = c[r] + bb;
            }
        }
#pragma unroll
        for (int s = 0; s < 4; ++s) bcur[s] = bnext[s];
    }
}

// ---------------- host side ----------------

static inline size_t alignUp(size_t v, size_t a) { return (v + a - 1) & ~(a - 1); }

extern "C" void kernel_launch(void* const* d_in, const int* in_sizes, int n_in,
                              void* d_out, int out_size, void* d_ws, size_t ws_size,
                              hipStream_t stream) {
    const float* x      = (const float*)d_in[0];   // [N,128]
    const int*   ei     = (const int*)  d_in[1];   // [2,E]
    const float* W_conv = (const float*)d_in[2];   // [128,128]
    const float* b_conv = (const float*)d_in[3];   // [128]
    const float* gamma  = (const float*)d_in[4];   // [128]
    const float* beta   = (const float*)d_in[5];   // [128]
    const float* W_fc   = (const float*)d_in[6];   // [128,40]
    const float* b_fc   = (const float*)d_in[7];   // [40]
    float* out = (float*)d_out;

    const int N = in_sizes[0] / CDIM;
    const int E = in_sizes[1] / 2;
    const int* srcp = ei;
    const int* dstp = ei + E;

    // workspace layout
    char* ws = (char*)d_ws;
    size_t off = 0;
    float* dinv = (float*)(ws + off); off = alignUp(off + (size_t)N * 4, 256);
    float* h    = (float*)(ws + off); off = alignUp(off + (size_t)N * CDIM * 4, 256);
    float* agg  = (float*)(ws + off); off = alignUp(off + (size_t)N * CDIM * 4, 256);
    __bf16* Wbf = (__bf16*)(ws + off); off = alignUp(off + (size_t)CDIM * CDIM * 2, 256);
    (void)ws_size; (void)n_in; (void)out_size;

    const int gN   = (N + 255) / 256;
    const int gE   = (E + 255) / 256;
    const int gRow = (N + 127) / 128;

    // 1-3: symmetric normalization factors
    k_deg_init<<<gN, 256, 0, stream>>>(dinv, N);
    k_deg_acc<<<gE, 256, 0, stream>>>(dstp, dinv, E);
    k_rsqrt_inplace<<<gN, 256, 0, stream>>>(dinv, N);

    // 4: W_conv -> bf16 fragment layout
    k_pack_wconv<<<(8 * 4 * 32 * 16 + 255) / 256, 256, 0, stream>>>(W_conv, Wbf);

    // 5: h = x @ W_conv (WMMA, double-buffered B)
    k_gemm_conv<<<gRow, 256, 0, stream>>>(x, Wbf, h, N);

    // 6: agg = b_conv + self-loop contribution
    k_agg_init<<<((size_t)N * (CDIM / 4) + 255) / 256, 256, 0, stream>>>(h, dinv, b_conv, agg, N);

    // 7: edge scatter with float atomics (L2-resident)
    k_scatter<<<(E + 7) / 8, 256, 0, stream>>>(srcp, dstp, h, dinv, agg, E);

    // 8: fused relu*x + LayerNorm + residual + (ln+x)@W_fc + b_fc (WMMA)
    k_ln_fc<<<gRow, 256, 0, stream>>>(agg, x, gamma, beta, W_fc, b_fc, out, N);
}
